// Pixel2Pointcloud_70806830842026
// MI455X (gfx1250) — matrix operand order repaired and checked by reading, not compiled, not run on hardware
//
#include <hip/hip_runtime.h>
#include <math.h>

typedef __attribute__((ext_vector_type(16))) _Float16 v16h;
typedef __attribute__((ext_vector_type(8)))  float    v8f;

// ---------------------------------------------------------------------------
// Tiled linear:  Y[r, ycol0+n] = act( X[r,0:Fin] @ W[n,0:Fin]^T + b[n] )
// Block = 128 threads (4 waves) computing a 64x64 output tile.
// Wave w owns row strip [16w,16w+16) and holds 4 accumulators over the 4
// column sub-tiles -> 4 v_wmma per K-chunk per wave.
// X/W tiles staged into LDS as f32 via GLOBAL_LOAD_ASYNC_TO_LDS_B32
// (ASYNCcnt), converted f32->f16 while building fragments.
// grid.x = 16384/64, grid.y = ceil(Fout/64).
// ---------------------------------------------------------------------------
__global__ __launch_bounds__(128) void linear_wmma(
    const float* __restrict__ X, int ldX, int Fin,
    const float* __restrict__ W, const float* __restrict__ Bias,
    float* __restrict__ Y, int ldY, int ycol0, int Fout, int do_relu)
{
  __shared__ float Xs[64][32];
  __shared__ float Ws[64][32];
  const int tid  = threadIdx.x;
  const int lane = tid & 31;
  const int wv   = tid >> 5;
  const int row0 = blockIdx.x * 64;
  const int n0   = blockIdx.y * 64;
  const int nj   = min(4, (Fout - n0 + 15) >> 4);   // uniform per block

  v8f acc0 = {}, acc1 = {}, acc2 = {}, acc3 = {};
  const int kTiles = (Fin + 31) >> 5;
  for (int kt = 0; kt < kTiles; ++kt) {
    const int k0 = kt << 5;
    // stage X tile (64 rows x 32 K) f32 -> LDS, async
    for (int i = tid; i < 64 * 32; i += 128) {
      int m = i >> 5, kk = i & 31;
      unsigned lo = (unsigned)(size_t)&Xs[m][kk];
      if (k0 + kk < Fin) {
        const float* g = &X[(size_t)(row0 + m) * ldX + k0 + kk];
        asm volatile("global_load_async_to_lds_b32 %0, %1, off"
                     :: "v"(lo), "v"(g) : "memory");
      } else {
        Xs[m][kk] = 0.f;
      }
    }
    // stage W tile (64 out-features x 32 K) f32 -> LDS, async
    for (int i = tid; i < 64 * 32; i += 128) {
      int n = i >> 5, kk = i & 31;
      int gn = n0 + n;
      unsigned lo = (unsigned)(size_t)&Ws[n][kk];
      if (gn < Fout && (k0 + kk) < Fin) {
        const float* g = &W[(size_t)gn * Fin + k0 + kk];
        asm volatile("global_load_async_to_lds_b32 %0, %1, off"
                     :: "v"(lo), "v"(g) : "memory");
      } else {
        Ws[n][kk] = 0.f;
      }
    }
    asm volatile("s_wait_asynccnt 0" ::: "memory");
    __syncthreads();

    // fragments per ISA wave32 16-bit layouts; convert f32->f16 on LDS read
    const int half = lane >> 4;
    const int ml   = lane & 15;
    v16h a;
    #pragma unroll
    for (int e = 0; e < 16; ++e) {
      int ka = ((e >> 3) << 4) + (half << 3) + (e & 7);  // A 16x32 K index
      a[e] = (_Float16)Xs[16 * wv + ml][ka];
    }
    #pragma unroll
    for (int j = 0; j < 4; ++j) {
      if (j < nj) {                                      // uniform branch
        v16h b;
        #pragma unroll
        for (int e = 0; e < 16; ++e) {
          int kb = ((e >> 3) << 4) + (half << 3) + (e & 7);  // B 32x16 K index
          b[e] = (_Float16)Ws[16 * j + ml][kb];
        }
        v8f c = (j == 0) ? acc0 : (j == 1) ? acc1 : (j == 2) ? acc2 : acc3;
        c = __builtin_amdgcn_wmma_f32_16x16x32_f16(false, a, false, b,
                                                   (short)0, c, false, false);
        if (j == 0) acc0 = c; else if (j == 1) acc1 = c;
        else if (j == 2) acc2 = c; else acc3 = c;
      }
    }
    __syncthreads();
  }
  // C layout: lane holds N=lane&15, VGPR r holds M = r + 8*(lane>>4)
  const int half = lane >> 4, nl = lane & 15;
  #pragma unroll
  for (int j = 0; j < 4; ++j) {
    if (j < nj) {
      int gn = n0 + 16 * j + nl;
      if (gn < Fout) {
        float bb = Bias ? Bias[gn] : 0.f;
        v8f c = (j == 0) ? acc0 : (j == 1) ? acc1 : (j == 2) ? acc2 : acc3;
        #pragma unroll
        for (int r = 0; r < 8; ++r) {
          int gm = row0 + 16 * wv + half * 8 + r;
          float v = c[r] + bb;
          if (do_relu) v = v > 0.f ? v : 0.f;
          Y[(size_t)gm * ldY + ycol0 + gn] = v;
        }
      }
    }
  }
}

// ---------------------------------------------------------------------------
// Implicit-GEMM conv + bias + relu.  64 co x 64 positions per block.
// Wave w owns position strip [16w,16w+16) (fixed B fragment) and loops over
// up to 4 co sub-tiles (A fragments) -> 4 accumulators, 4 v_wmma per K-chunk.
// grid.x = ceil(Cout/64), grid.y = ceil(B*OH*OW/64). block = 128.
// ---------------------------------------------------------------------------
__global__ __launch_bounds__(128) void conv_wmma(
    const float* __restrict__ Xin, const float* __restrict__ Wt,
    const float* __restrict__ Bias, float* __restrict__ Yout,
    int Bn, int Cin, int IH, int IW, int Cout, int OH, int OW,
    int K, int Str, int Pad)
{
  __shared__ _Float16 Wts[64][32];
  __shared__ _Float16 Pts[64][32];
  const int tid = threadIdx.x, lane = tid & 31, wv = tid >> 5;
  const int co0 = blockIdx.x * 64;
  const int p0  = blockIdx.y * 64;
  const int P    = Bn * OH * OW;
  const int Ktot = Cin * K * K;
  const int KK   = K * K;
  const int nj   = min(4, (Cout - co0 + 15) >> 4);   // uniform per block

  v8f acc0 = {}, acc1 = {}, acc2 = {}, acc3 = {};
  const int kTiles = (Ktot + 31) >> 5;
  for (int kt = 0; kt < kTiles; ++kt) {
    const int k0 = kt << 5;
    if (k0 + 32 < Ktot) {   // uniform: hint next weight tile (global_prefetch_b8)
      __builtin_prefetch(&Wt[(size_t)(co0 + (tid & 63)) * Ktot + k0 + 32], 0, 1);
    }
    // weights tile 64x32, f32 -> f16
    for (int i = tid; i < 64 * 32; i += 128) {
      int m = i >> 5, kk = i & 31;
      int gm = co0 + m, gk = k0 + kk;
      float v = (gm < Cout && gk < Ktot) ? Wt[(size_t)gm * Ktot + gk] : 0.f;
      Wts[m][kk] = (_Float16)v;
    }
    // im2col patch tile 64 positions x 32 K, f32 -> f16
    for (int i = tid; i < 64 * 32; i += 128) {
      int p = i >> 5, kk = i & 31;
      int gp = p0 + p, gk = k0 + kk;
      float v = 0.f;
      if (gp < P && gk < Ktot) {
        int b  = gp / (OH * OW); int rem = gp - b * (OH * OW);
        int oh = rem / OW;       int ow  = rem - oh * OW;
        int ci = gk / KK;        int r2  = gk - ci * KK;
        int kh = r2 / K;         int kw  = r2 - kh * K;
        int ih = oh * Str - Pad + kh;
        int iw = ow * Str - Pad + kw;
        if (ih >= 0 && ih < IH && iw >= 0 && iw < IW)
          v = Xin[(((size_t)b * Cin + ci) * IH + ih) * IW + iw];
      }
      Pts[p][kk] = (_Float16)v;
    }
    __syncthreads();
    const int half = lane >> 4, ml = lane & 15;
    v16h b;
    #pragma unroll
    for (int e = 0; e < 16; ++e)
      b[e] = Pts[16 * wv + ml][(half << 4) + e];
    #pragma unroll
    for (int j = 0; j < 4; ++j) {
      if (j < nj) {                                    // uniform branch
        v16h a;
        #pragma unroll
        for (int e = 0; e < 16; ++e) {
          int ka = (half << 4) + e;
          a[e] = Wts[16 * j + ml][ka];
        }
        v8f c = (j == 0) ? acc0 : (j == 1) ? acc1 : (j == 2) ? acc2 : acc3;
        c = __builtin_amdgcn_wmma_f32_16x16x32_f16(false, a, false, b,
                                                   (short)0, c, false, false);
        if (j == 0) acc0 = c; else if (j == 1) acc1 = c;
        else if (j == 2) acc2 = c; else acc3 = c;
      }
    }
    __syncthreads();
  }
  const int half = lane >> 4, nl = lane & 15;
  const int gp = p0 + 16 * wv + nl;
  if (gp < P) {
    int b  = gp / (OH * OW); int rem = gp - b * (OH * OW);
    int oh = rem / OW;       int ow  = rem - oh * OW;
    #pragma unroll
    for (int j = 0; j < 4; ++j) {
      if (j < nj) {
        v8f c = (j == 0) ? acc0 : (j == 1) ? acc1 : (j == 2) ? acc2 : acc3;
        #pragma unroll
        for (int r = 0; r < 8; ++r) {
          int co = co0 + 16 * j + half * 8 + r;
          if (co < Cout) {
            float v = c[r] + Bias[co];
            v = v > 0.f ? v : 0.f;
            Yout[(((size_t)b * Cout + co) * OH + oh) * OW + ow] = v;
          }
        }
      }
    }
  }
}

// ---------------------------------------------------------------------------
// Per (b,c) spatial mean/var (ddof=1) of an NCHW feature map.
// ---------------------------------------------------------------------------
__global__ void img_stats(const float* __restrict__ F, int HW,
                          float* __restrict__ mean, float* __restrict__ var)
{
  const float* p = F + (size_t)blockIdx.x * HW;
  float s = 0.f, ss = 0.f;
  for (int i = threadIdx.x; i < HW; i += blockDim.x) { float v = p[i]; s += v; ss += v * v; }
  __shared__ float sh1[256], sh2[256];
  sh1[threadIdx.x] = s; sh2[threadIdx.x] = ss;
  __syncthreads();
  for (int off = blockDim.x / 2; off > 0; off >>= 1) {
    if (threadIdx.x < off) { sh1[threadIdx.x] += sh1[threadIdx.x + off];
                             sh2[threadIdx.x] += sh2[threadIdx.x + off]; }
    __syncthreads();
  }
  if (threadIdx.x == 0) {
    float mu = sh1[0] / (float)HW;
    float vv = (sh2[0] - (float)HW * mu * mu) / (float)(HW - 1);
    mean[blockIdx.x] = mu; var[blockIdx.x] = vv;
  }
}

// ---------------------------------------------------------------------------
// AdaIN: per-row mean/var (ddof=1) via wave32 shuffle reduction, then
// out = ((x-mu)/sqrt(v+eps) + imean[b,c]) * sqrt(ivar[b,c]+eps).
// ---------------------------------------------------------------------------
__global__ __launch_bounds__(128) void adain_norm(
    const float* __restrict__ Xpc, int F,
    const float* __restrict__ imean, const float* __restrict__ ivar,
    int N, float* __restrict__ Out)
{
  const int lane = threadIdx.x & 31, wv = threadIdx.x >> 5;
  const int row  = blockIdx.x * 4 + wv;
  const int b    = row / N;
  const float* x = Xpc + (size_t)row * F;
  float s = 0.f, ss = 0.f;
  for (int i = lane; i < F; i += 32) { float v = x[i]; s += v; ss += v * v; }
  #pragma unroll
  for (int m = 16; m > 0; m >>= 1) { s += __shfl_xor(s, m, 32); ss += __shfl_xor(ss, m, 32); }
  float mu   = s / (float)F;
  float var  = (ss - (float)F * mu * mu) / (float)(F - 1);
  float rinv = 1.0f / sqrtf(var + 1e-8f);
  float* o = Out + (size_t)row * F;
  for (int i = lane; i < F; i += 32) {
    float nrm = (x[i] - mu) * rinv;
    o[i] = (nrm + imean[b * F + i]) * sqrtf(ivar[b * F + i] + 1e-8f);
  }
}

// ---------------------------------------------------------------------------
// Bilinear projection sampler (faithful floor/ceil weighting).
// ---------------------------------------------------------------------------
__global__ void project_bilinear(
    const float* __restrict__ F, int C, int H, int W,
    const float* __restrict__ pc, float* __restrict__ Xcat,
    int ldc, int col0, int N, int total)
{
  int r = blockIdx.x * blockDim.x + threadIdx.x;
  if (r >= total) return;
  int b = r / N;
  float Xc = pc[(size_t)r * 3 + 0];
  float Yc = pc[(size_t)r * 3 + 1];
  float Zc = pc[(size_t)r * 3 + 2];
  float h = fminf(fmaxf(248.f * Yc / Zc + 111.5f, 0.f), 223.f);
  float w = fminf(fmaxf(248.f * (-Xc) / Zc + 111.5f, 0.f), 223.f);
  float x = h / (223.f / ((float)H - 1.f));
  float y = w / (223.f / ((float)W - 1.f));
  float x1 = floorf(x), y1 = floorf(y);
  float x2 = ceilf(x),  y2 = ceilf(y);
  int xi1 = (int)x1, yi1 = (int)y1, xi2 = (int)x2, yi2 = (int)y2;
  float w11 = (x2 - x) * (y2 - y), w21 = (x - x1) * (y2 - y);
  float w12 = (x2 - x) * (y - y1), w22 = (x - x1) * (y - y1);
  const float* Fb = F + (size_t)b * C * H * W;
  float* o = Xcat + (size_t)r * ldc + col0;
  for (int c = 0; c < C; ++c) {
    const float* Fc = Fb + (size_t)c * H * W;
    float q11 = Fc[xi1 * W + yi1], q21 = Fc[xi2 * W + yi1];
    float q12 = Fc[xi1 * W + yi2], q22 = Fc[xi2 * W + yi2];
    o[c] = q11 * w11 + q21 * w21 + q12 * w12 + q22 * w22;
  }
}

__global__ void copy_initpc(const float* __restrict__ pc, float* __restrict__ Xcat,
                            int ldc, int col0, int total)
{
  int i = blockIdx.x * blockDim.x + threadIdx.x;
  if (i < total) {
    int r = i / 3, j = i - r * 3;
    Xcat[(size_t)r * ldc + col0 + j] = pc[i];
  }
}

// ---------------------------------------------------------------------------
// Host-side orchestration
// ---------------------------------------------------------------------------
static void launch_linear(hipStream_t s, const float* X, int ldX, int Fin,
                          const float* W, const float* b,
                          float* Y, int ldY, int ycol0, int Fout, int relu)
{
  dim3 grid(16384 / 64, (Fout + 63) / 64);
  linear_wmma<<<grid, 128, 0, s>>>(X, ldX, Fin, W, b, Y, ldY, ycol0, Fout, relu);
}

static void launch_conv(hipStream_t s, const float* x, const float* w, const float* b,
                        float* y, int B, int Cin, int IH, int IW,
                        int Cout, int OH, int OW, int K, int Str, int Pad)
{
  int P = B * OH * OW;
  dim3 grid((Cout + 63) / 64, (P + 63) / 64);
  conv_wmma<<<grid, 128, 0, s>>>(x, w, b, y, B, Cin, IH, IW, Cout, OH, OW, K, Str, Pad);
}

extern "C" void kernel_launch(void* const* d_in, const int* in_sizes, int n_in,
                              void* d_out, int out_size, void* d_ws, size_t ws_size,
                              hipStream_t stream)
{
  (void)in_sizes; (void)n_in; (void)out_size; (void)ws_size;
  // Pytree leaf order: image, init_pc, then params dict keys sorted:
  // 'dec', 'enc', 'fcs', 'pc_blocks'.
  const float* image  = (const float*)d_in[0];
  const float* initpc = (const float*)d_in[1];
  const float* decw[4], *decb[4];
  for (int i = 0; i < 4; ++i) { decw[i] = (const float*)d_in[2 + 2 * i];
                                decb[i] = (const float*)d_in[3 + 2 * i]; }
  const float* cw[18], *cb[18];
  for (int i = 0; i < 18; ++i) { cw[i] = (const float*)d_in[10 + 2 * i];
                                 cb[i] = (const float*)d_in[11 + 2 * i]; }
  const float* fcw[4], *fcb[4];
  for (int i = 0; i < 4; ++i) { fcw[i] = (const float*)d_in[46 + 2 * i];
                                fcb[i] = (const float*)d_in[47 + 2 * i]; }
  const float* p1w[4], *p1b[4], *p2w[4], *p2b[4];
  for (int i = 0; i < 4; ++i) {
    p1w[i] = (const float*)d_in[54 + 4 * i]; p1b[i] = (const float*)d_in[55 + 4 * i];
    p2w[i] = (const float*)d_in[56 + 4 * i]; p2b[i] = (const float*)d_in[57 + 4 * i];
  }

  // workspace layout (floats)
  float* ws    = (float*)d_ws;
  float* convA = ws;                      // 8*16*224*224
  float* convB = convA + 6422528;
  float* f1    = convB + 6422528;         // 8*64*56*56
  float* f2    = f1 + 1605632;            // 8*128*28*28
  float* f3    = f2 + 802816;             // 8*256*14*14
  float* f4    = f3 + 401408;             // 8*512*7*7
  float* imean = f4 + 200704;             // 8*960
  float* ivar  = imean + 7680;
  float* pcA   = ivar + 7680;             // 16384*512
  float* pcB   = pcA + 8388608;
  float* adT   = pcB + 8388608;
  float* Xcat  = adT + 8388608;           // 16384*1923

  // ---- image encoder ----
  launch_conv(stream, image, cw[0],  cb[0],  convA, 8,   3, 224, 224,  16, 224, 224, 3, 1, 1);
  launch_conv(stream, convA, cw[1],  cb[1],  convB, 8,  16, 224, 224,  16, 224, 224, 3, 1, 1);
  launch_conv(stream, convB, cw[2],  cb[2],  convA, 8,  16, 224, 224,  32, 112, 112, 3, 2, 1);
  launch_conv(stream, convA, cw[3],  cb[3],  convB, 8,  32, 112, 112,  32, 112, 112, 3, 1, 1);
  launch_conv(stream, convB, cw[4],  cb[4],  convA, 8,  32, 112, 112,  32, 112, 112, 3, 1, 1);
  launch_conv(stream, convA, cw[5],  cb[5],  convB, 8,  32, 112, 112,  64,  56,  56, 3, 2, 1);
  launch_conv(stream, convB, cw[6],  cb[6],  convA, 8,  64,  56,  56,  64,  56,  56, 3, 1, 1);
  launch_conv(stream, convA, cw[7],  cb[7],  f1,    8,  64,  56,  56,  64,  56,  56, 3, 1, 1);
  launch_conv(stream, f1,    cw[8],  cb[8],  convA, 8,  64,  56,  56, 128,  28,  28, 3, 2, 1);
  launch_conv(stream, convA, cw[9],  cb[9],  convB, 8, 128,  28,  28, 128,  28,  28, 3, 1, 1);
  launch_conv(stream, convB, cw[10], cb[10], f2,    8, 128,  28,  28, 128,  28,  28, 3, 1, 1);
  launch_conv(stream, f2,    cw[11], cb[11], convA, 8, 128,  28,  28, 256,  14,  14, 5, 2, 2);
  launch_conv(stream, convA, cw[12], cb[12], convB, 8, 256,  14,  14, 256,  14,  14, 3, 1, 1);
  launch_conv(stream, convB, cw[13], cb[13], f3,    8, 256,  14,  14, 256,  14,  14, 3, 1, 1);
  launch_conv(stream, f3,    cw[14], cb[14], convA, 8, 256,  14,  14, 512,   7,   7, 5, 2, 2);
  launch_conv(stream, convA, cw[15], cb[15], convB, 8, 512,   7,   7, 512,   7,   7, 3, 1, 1);
  launch_conv(stream, convB, cw[16], cb[16], convA, 8, 512,   7,   7, 512,   7,   7, 3, 1, 1);
  launch_conv(stream, convA, cw[17], cb[17], f4,    8, 512,   7,   7, 512,   7,   7, 3, 1, 1);

  const int   Cs[4]  = {64, 128, 256, 512};
  const int   Hs[4]  = {56, 28, 14, 7};
  const int   pre[4] = {0, 64, 192, 448};
  const float* feats[4] = {f1, f2, f3, f4};
  for (int i = 0; i < 4; ++i)
    img_stats<<<8 * Cs[i], 256, 0, stream>>>(feats[i], Hs[i] * Hs[i],
                                             imean + 8 * pre[i], ivar + 8 * pre[i]);

  // ---- point-cloud MLP chain + AdaIN + fc + projections into Xcat ----
  const int adOff[4] = {0, 64, 192, 448};
  const int prOff[4] = {960, 1024, 1152, 1408};
  const float* cur = initpc; int curF = 3;
  for (int i = 0; i < 4; ++i) {
    int f = Cs[i];
    launch_linear(stream, cur, curF, curF, p1w[i], p1b[i], pcB, f, 0, f, 1);
    launch_linear(stream, pcB, f, f,       p2w[i], p2b[i], pcA, f, 0, f, 1);
    adain_norm<<<16384 / 4, 128, 0, stream>>>(pcA, f, imean + 8 * pre[i],
                                              ivar + 8 * pre[i], 2048, adT);
    launch_linear(stream, adT, f, f, fcw[i], fcb[i], Xcat, 1923, adOff[i], f, 0);
    project_bilinear<<<(16384 + 255) / 256, 256, 0, stream>>>(
        feats[i], Cs[i], Hs[i], Hs[i], initpc, Xcat, 1923, prOff[i], 2048, 16384);
    cur = pcA; curF = f;
  }
  copy_initpc<<<(49152 + 255) / 256, 256, 0, stream>>>(initpc, Xcat, 1923, 1920, 49152);

  // ---- decoder ----
  launch_linear(stream, Xcat, 1923, 1923, decw[0], decb[0], pcB, 512, 0, 512, 1);
  launch_linear(stream, pcB,  512,  512,  decw[1], decb[1], pcA, 256, 0, 256, 1);
  launch_linear(stream, pcA,  256,  256,  decw[2], decb[2], pcB, 128, 0, 128, 1);
  launch_linear(stream, pcB,  128,  128,  decw[3], decb[3], (float*)d_out, 3, 0, 3, 1);
}
